// MulitHeadAttentionLayer_64802466562068
// MI455X (gfx1250) — matrix operand order repaired and checked by reading, not compiled, not run on hardware
//
#include <hip/hip_runtime.h>
#include <hip/hip_bf16.h>

// ---------------------------------------------------------------------------
// MultiHeadAttentionLayer for MI455X (gfx1250, wave32, WMMA)
//   B=2, C=128, T=4,H=32,W=32 -> N=4096 tokens, L=6 layers
//   scores softmax is over axis=1 (query axis) => COLUMN-wise softmax.
//   Two-pass scheme: pass1 = column stats (max, sumexp), pass2 = P @ G1.
//   All GEMMs use v_wmma_f32_16x16x32_bf16; raw v_exp_f32 for softmax.
// ---------------------------------------------------------------------------

#define Bsz 2
#define Cdim 128
#define Ntok 4096
#define Lnum 6
// inv_sqrt(N) * log2(e): work in base-2 exponentials throughout
#define SCALE2 (0.015625f * 1.44269504088896340736f)

typedef __attribute__((ext_vector_type(16))) __bf16 bf16x16;
typedef __attribute__((ext_vector_type(8)))  float  f32x8;
typedef __attribute__((ext_vector_type(4)))  int    i32x4;
typedef __attribute__((ext_vector_type(8)))  int    i32x8;

static __device__ __forceinline__ unsigned short f32_to_bf16_rne(float f) {
    unsigned int u = __float_as_uint(f);
    u += 0x7FFFu + ((u >> 16) & 1u);       // round-to-nearest-even
    return (unsigned short)(u >> 16);
}

// bare v_exp_f32 (2^x), denorm-flushing fast path: fine for softmax weights
static __device__ __forceinline__ float fast_exp2(float x) {
    return __builtin_amdgcn_exp2f(x);
}

static __device__ __forceinline__ f32x8 wmma_bf16(bf16x16 a, bf16x16 b, f32x8 c) {
    return __builtin_amdgcn_wmma_f32_16x16x32_bf16(false, a, false, b, (short)0, c, false, false);
}

// A-fragment (16x32, bf16) from row-major [token][channel=128] bf16 array.
// Lane holds row (lane&15); contiguous K: k = g*16 .. g*16+15.
static __device__ __forceinline__ bf16x16 load_a_frag(const unsigned short* base,
                                                      int row0, int kbase, int lane) {
    const unsigned short* p = base + (size_t)(row0 + (lane & 15)) * Cdim + kbase;
    int g8 = (lane >> 4) * 16;
    i32x4 lo = *(const i32x4*)(p + g8);
    i32x4 hi = *(const i32x4*)(p + 8 + g8);
    i32x8 r;
    r[0] = lo[0]; r[1] = lo[1]; r[2] = lo[2]; r[3] = lo[3];
    r[4] = hi[0]; r[5] = hi[1]; r[6] = hi[2]; r[7] = hi[3];
    return __builtin_bit_cast(bf16x16, r);
}

// B-fragment (32x16, bf16): column n' of B = row (row0+n') of the array,
// contiguous along K. Lane reads 32B at k-offset g*16.
static __device__ __forceinline__ bf16x16 load_b_frag(const unsigned short* base,
                                                      int row0, int stride, int koff, int lane) {
    const unsigned short* p = base + (size_t)(row0 + (lane & 15)) * stride
                                   + koff + (lane >> 4) * 16;
    return __builtin_bit_cast(bf16x16, *(const i32x8*)p);
}

// ---------------------------------------------------------------------------
// Kernel 1: projections. F1[n][c], F2[n][c] token-major; G1t[c][n] channel-major.
// grid = 64 token-tiles * L * B blocks, 256 threads.
// ---------------------------------------------------------------------------
__global__ __launch_bounds__(256) void proj_kernel(
                            const float* __restrict__ x,
                            const float* __restrict__ W1, const float* __restrict__ b1,
                            const float* __restrict__ W2, const float* __restrict__ b2,
                            const float* __restrict__ Wg, const float* __restrict__ bg,
                            unsigned short* __restrict__ F1,
                            unsigned short* __restrict__ F2,
                            unsigned short* __restrict__ G1t) {
    __shared__ float xs[Cdim][64];
    __shared__ float wsm[Cdim][Cdim];
    __shared__ float bsh[Cdim];
    const int tid  = threadIdx.x;
    const int tile = blockIdx.x & 63;
    const int bl   = blockIdx.x >> 6;
    const int b    = bl & 1, l = bl >> 1;
    const int n0   = tile * 64;

    for (int idx = tid; idx < Cdim * 64; idx += 256) {
        int c = idx >> 6, j = idx & 63;
        xs[c][j] = x[((size_t)(b * Cdim + c)) * Ntok + n0 + j];
    }
    const float* Ws[3] = { W1 + l * Cdim * Cdim, W2 + l * Cdim * Cdim, Wg + l * Cdim * Cdim };
    const float* bs[3] = { b1 + l * Cdim,        b2 + l * Cdim,        bg + l * Cdim };
    const int tok = tid & 63, og = tid >> 6;
    const size_t blo = (size_t)(l * Bsz + b);

    for (int p = 0; p < 3; ++p) {
        __syncthreads();
        for (int idx = tid; idx < Cdim * Cdim; idx += 256)
            wsm[idx >> 7][idx & 127] = Ws[p][idx];
        if (tid < Cdim) bsh[tid] = bs[p][tid];
        __syncthreads();
        for (int oo = 0; oo < 32; ++oo) {
            int o = og * 32 + oo;
            float a = bsh[o];
            #pragma unroll 4
            for (int c = 0; c < Cdim; ++c) a = fmaf(wsm[o][c], xs[c][tok], a);
            unsigned short h = f32_to_bf16_rne(a);
            if (p == 0)      F1[(blo * Ntok + n0 + tok) * Cdim + o] = h;
            else if (p == 1) F2[(blo * Ntok + n0 + tok) * Cdim + o] = h;
            else             G1t[(blo * Cdim + o) * Ntok + n0 + tok] = h;
        }
    }
}

// ---------------------------------------------------------------------------
// Kernel 2 (pass 1): column-softmax stats. One wave owns 32 columns (2 tiles),
// streams all 4096 rows in 16-row WMMA tiles. Row loop unrolled x2 with
// ping-pong A-fragment buffers (no rotating copies, no spills thanks to
// launch_bounds). total waves = L*B*128; block = 4 waves.
// ---------------------------------------------------------------------------
__global__ __launch_bounds__(128, 1) void pass1_stats(
                            const unsigned short* __restrict__ F1,
                            const unsigned short* __restrict__ F2,
                            float* __restrict__ cmax, float* __restrict__ crcp) {
    const int lane = threadIdx.x & 31;
    const int wid  = (int)((blockIdx.x * blockDim.x + threadIdx.x) >> 5);
    const int slot = wid & 127;           // 128 slots of 32 columns
    const int bl   = wid >> 7;
    const int b    = bl & 1, l = bl >> 1;
    const int m0   = slot * 32;
    const size_t blo = (size_t)(l * Bsz + b);
    const unsigned short* f1 = F1 + blo * Ntok * Cdim;
    const unsigned short* f2 = F2 + blo * Ntok * Cdim;

    bf16x16 bfrag[2][4];
    #pragma unroll
    for (int ct = 0; ct < 2; ++ct)
        #pragma unroll
        for (int kc = 0; kc < 4; ++kc)
            bfrag[ct][kc] = load_b_frag(f2, m0 + ct * 16, Cdim, kc * 32, lane);

    float mrun[2] = { -__builtin_inff(), -__builtin_inff() };
    float srun[2] = { 0.f, 0.f };

    // process one 16-row tile of scores against both column tiles
    auto tile_update = [&](const bf16x16 (&af)[4]) {
        #pragma unroll
        for (int ct = 0; ct < 2; ++ct) {
            f32x8 s = { 0.f, 0.f, 0.f, 0.f, 0.f, 0.f, 0.f, 0.f };
            #pragma unroll
            for (int kc = 0; kc < 4; ++kc) s = wmma_bf16(af[kc], bfrag[ct][kc], s);
            float v[8], tm = -__builtin_inff();
            #pragma unroll
            for (int j = 0; j < 8; ++j) { v[j] = s[j] * SCALE2; tm = fmaxf(tm, v[j]); }
            float nm  = fmaxf(mrun[ct], tm);
            float acc = srun[ct] * fast_exp2(mrun[ct] - nm);
            #pragma unroll
            for (int j = 0; j < 8; ++j) acc += fast_exp2(v[j] - nm);
            mrun[ct] = nm; srun[ct] = acc;
        }
    };

    bf16x16 a0[4], a1[4];
    #pragma unroll
    for (int kc = 0; kc < 4; ++kc) a0[kc] = load_a_frag(f1, 0, kc * 32, lane);

    for (int r0 = 0; r0 < Ntok; r0 += 32) {            // 128 iterations, even
        #pragma unroll
        for (int kc = 0; kc < 4; ++kc) a1[kc] = load_a_frag(f1, r0 + 16, kc * 32, lane);
        tile_update(a0);
        if (r0 + 32 < Ntok) {
            #pragma unroll
            for (int kc = 0; kc < 4; ++kc) a0[kc] = load_a_frag(f1, r0 + 32, kc * 32, lane);
        }
        tile_update(a1);
    }
    #pragma unroll
    for (int ct = 0; ct < 2; ++ct) {
        float mo = __shfl_xor(mrun[ct], 16, 32);
        float so = __shfl_xor(srun[ct], 16, 32);
        float mt = fmaxf(mrun[ct], mo);
        float st = srun[ct] * fast_exp2(mrun[ct] - mt) + so * fast_exp2(mo - mt);
        if (lane < 16) {
            size_t off = blo * Ntok + m0 + ct * 16 + lane;
            cmax[off] = mt;
            crcp[off] = 1.0f / st;
        }
    }
}

// ---------------------------------------------------------------------------
// Kernel 3 (pass 2): recompute S tiles, P = 2^(s-cmax)*crcp, bounce P through
// LDS into A-fragment layout, multiply against channel-major G1 with WMMA.
// All global loads for an iteration issue up front so G1 fetches overlap the
// score WMMAs and the exp/LDS bounce. One wave owns a 16-row output tile.
// total waves = L*B*256; block = 4 waves.
// ---------------------------------------------------------------------------
__global__ __launch_bounds__(128, 1) void pass2_attn(
                           const unsigned short* __restrict__ F1,
                           const unsigned short* __restrict__ F2,
                           const unsigned short* __restrict__ G1t,
                           const float* __restrict__ cmax,
                           const float* __restrict__ crcp,
                           float* __restrict__ acc) {
    __shared__ __align__(16) unsigned short pbuf[4][16][32];
    const int lane  = threadIdx.x & 31;
    const int w     = threadIdx.x >> 5;
    const int wid   = (int)(blockIdx.x * (blockDim.x >> 5) + w);
    const int rslot = wid & 255;
    const int bl    = wid >> 8;
    const int b     = bl & 1, l = bl >> 1;
    const int r0    = rslot * 16;
    const size_t blo = (size_t)(l * Bsz + b);
    const unsigned short* f1 = F1  + blo * Ntok * Cdim;
    const unsigned short* f2 = F2  + blo * Ntok * Cdim;
    const unsigned short* g1 = G1t + blo * Ntok * Cdim;
    const float* cm = cmax + blo * Ntok;
    const float* cr = crcp + blo * Ntok;
    const int col = lane & 15, g = lane >> 4;

    bf16x16 afrag[4];
    #pragma unroll
    for (int kc = 0; kc < 4; ++kc) afrag[kc] = load_a_frag(f1, r0, kc * 32, lane);

    f32x8 oacc[8];
    #pragma unroll
    for (int ci = 0; ci < 8; ++ci) oacc[ci] = (f32x8){0.f,0.f,0.f,0.f,0.f,0.f,0.f,0.f};

    for (int m0 = 0; m0 < Ntok; m0 += 32) {
        // issue every global load for this iteration up front
        bf16x16 sb[2][4];
        #pragma unroll
        for (int ct = 0; ct < 2; ++ct)
            #pragma unroll
            for (int kc = 0; kc < 4; ++kc)
                sb[ct][kc] = load_b_frag(f2, m0 + ct * 16, Cdim, kc * 32, lane);
        bf16x16 gf[8];
        #pragma unroll
        for (int ci = 0; ci < 8; ++ci)
            gf[ci] = load_b_frag(g1, ci * 16, Ntok, m0, lane);
        float cmv0 = cm[m0 + col],      crv0 = cr[m0 + col];
        float cmv1 = cm[m0 + 16 + col], crv1 = cr[m0 + 16 + col];

        // scores for 32 columns, then P -> LDS (bf16, row-major 16x32)
        #pragma unroll
        for (int ct = 0; ct < 2; ++ct) {
            f32x8 s = { 0.f, 0.f, 0.f, 0.f, 0.f, 0.f, 0.f, 0.f };
            #pragma unroll
            for (int kc = 0; kc < 4; ++kc) s = wmma_bf16(afrag[kc], sb[ct][kc], s);
            float cmv = ct ? cmv1 : cmv0;
            float crv = ct ? crv1 : crv0;
            #pragma unroll
            for (int j = 0; j < 8; ++j) {
                float p = fast_exp2(s[j] * SCALE2 - cmv) * crv;
                pbuf[w][j + 8 * g][ct * 16 + col] = f32_to_bf16_rne(p);
            }
        }
        asm volatile("s_wait_dscnt 0" ::: "memory");   // cross-lane LDS visibility

        bf16x16 pfrag;
        {
            const unsigned short* p = &pbuf[w][col][0];
            i32x4 lo = *(const i32x4*)(p + g * 16);
            i32x4 hi = *(const i32x4*)(p + 8 + g * 16);
            i32x8 r;
            r[0] = lo[0]; r[1] = lo[1]; r[2] = lo[2]; r[3] = lo[3];
            r[4] = hi[0]; r[5] = hi[1]; r[6] = hi[2]; r[7] = hi[3];
            pfrag = __builtin_bit_cast(bf16x16, r);
        }
        // P (16x32) @ G1 (32 x 128): B columns = G1t rows (channel-major, contiguous m)
        #pragma unroll
        for (int ci = 0; ci < 8; ++ci)
            oacc[ci] = wmma_bf16(pfrag, gf[ci], oacc[ci]);
    }
    float* ap = acc + blo * Ntok * Cdim;
    #pragma unroll
    for (int ci = 0; ci < 8; ++ci)
        #pragma unroll
        for (int j = 0; j < 8; ++j)
            ap[(size_t)(r0 + j + 8 * g) * Cdim + ci * 16 + col] = oacc[ci][j];
}

// ---------------------------------------------------------------------------
// Kernel 4: out[b][c][n] = x[b][c][n] + mean_l acc[l][b][n][c]
// ---------------------------------------------------------------------------
__global__ __launch_bounds__(256) void finalize_kernel(
                                const float* __restrict__ x,
                                const float* __restrict__ acc,
                                float* __restrict__ out) {
    size_t i = (size_t)blockIdx.x * 256 + threadIdx.x;   // over B*C*N
    int n = (int)(i & (Ntok - 1));
    size_t t = i >> 12;
    int c = (int)(t & (Cdim - 1));
    int b = (int)(t >> 7);
    float a = 0.f;
    #pragma unroll
    for (int l = 0; l < Lnum; ++l)
        a += acc[(((size_t)(l * Bsz + b)) * Ntok + n) * Cdim + c];
    out[i] = x[i] + a * (1.0f / Lnum);
}

// ---------------------------------------------------------------------------
extern "C" void kernel_launch(void* const* d_in, const int* in_sizes, int n_in,
                              void* d_out, int out_size, void* d_ws, size_t ws_size,
                              hipStream_t stream) {
    const float* x  = (const float*)d_in[0];
    const float* W1 = (const float*)d_in[1];
    const float* b1 = (const float*)d_in[2];
    const float* W2 = (const float*)d_in[3];
    const float* b2 = (const float*)d_in[4];
    const float* Wg = (const float*)d_in[5];
    const float* bg = (const float*)d_in[6];
    float* out = (float*)d_out;

    // workspace layout (~61 MB total)
    char* ws = (char*)d_ws;
    const size_t szF = (size_t)Lnum * Bsz * Ntok * Cdim * sizeof(unsigned short); // 12 MB
    unsigned short* F1  = (unsigned short*)(ws);
    unsigned short* F2  = (unsigned short*)(ws + szF);
    unsigned short* G1t = (unsigned short*)(ws + 2 * szF);
    float* cmax = (float*)(ws + 3 * szF);
    float* crcp = cmax + (size_t)Lnum * Bsz * Ntok;
    float* acc  = crcp + (size_t)Lnum * Bsz * Ntok;

    // 1) projections -> bf16 F1/F2 (token-major), G1t (channel-major)
    hipLaunchKernelGGL(proj_kernel, dim3(64 * Lnum * Bsz), dim3(256), 0, stream,
                       x, W1, b1, W2, b2, Wg, bg, F1, F2, G1t);
    // 2) column softmax stats: L*B*128 waves, 4 waves/block
    hipLaunchKernelGGL(pass1_stats, dim3((Lnum * Bsz * 128) / 4), dim3(128), 0, stream,
                       F1, F2, cmax, crcp);
    // 3) attention: L*B*256 waves, 4 waves/block
    hipLaunchKernelGGL(pass2_attn, dim3((Lnum * Bsz * 256) / 4), dim3(128), 0, stream,
                       F1, F2, G1t, cmax, crcp, acc);
    // 4) mean over layers + residual + transpose to [B,C,N]
    hipLaunchKernelGGL(finalize_kernel, dim3((Bsz * Cdim * Ntok) / 256), dim3(256), 0, stream,
                       x, acc, out);
}
